// MemoryEfficientGNN_54778012893479
// MI455X (gfx1250) — matrix operand order repaired and checked by reading, not compiled, not run on hardware
//
#include <hip/hip_runtime.h>
#include <hip/hip_bf16.h>

typedef __attribute__((ext_vector_type(16))) _Float16 v16h;
typedef __attribute__((ext_vector_type(8)))  _Float16 v8h;
typedef __attribute__((ext_vector_type(4)))  _Float16 v4h;
typedef __attribute__((ext_vector_type(8)))  float    v8f;

#define HIDN 128
#define FCD  64
#define SCB  256   // scan block size

// ---------------- f32 -> f16 convert ----------------
__global__ void k_f32_to_f16(const float* __restrict__ src, _Float16* __restrict__ dst, int n)
{
    int i = blockIdx.x * blockDim.x + threadIdx.x;
    if (i < n) dst[i] = (_Float16)src[i];
}

// ---------------- in-degree histogram (real edges only) ----------------
__global__ void k_count(const int* __restrict__ col, int* __restrict__ ecnt, int E)
{
    int i = blockIdx.x * blockDim.x + threadIdx.x;
    if (i < E) atomicAdd(&ecnt[col[i]], 1);
}

// dinv[v] = rsqrt(in_deg + 1 self-loop)
__global__ void k_dinv(const int* __restrict__ ecnt, float* __restrict__ dinv, int n)
{
    int i = blockIdx.x * blockDim.x + threadIdx.x;
    if (i < n) dinv[i] = rsqrtf((float)ecnt[i] + 1.0f);
}

// ---------------- two-level inclusive scan: ecnt -> rowptr[1..n] ----------------
__global__ void k_scan_local(const int* __restrict__ in, int* __restrict__ out,
                             int* __restrict__ bsum, int n)
{
    __shared__ int s[SCB];
    const int i = blockIdx.x * SCB + threadIdx.x;
    s[threadIdx.x] = (i < n) ? in[i] : 0;
    __syncthreads();
#pragma unroll
    for (int o = 1; o < SCB; o <<= 1) {
        int t = (threadIdx.x >= o) ? s[threadIdx.x - o] : 0;
        __syncthreads();
        s[threadIdx.x] += t;
        __syncthreads();
    }
    if (i < n) out[i] = s[threadIdx.x];
    if (threadIdx.x == SCB - 1) bsum[blockIdx.x] = s[SCB - 1];
}

__global__ void k_scan_bsum(int* __restrict__ bsum, int nb)   // single block, nb <= 512
{
    __shared__ int s[512];
    s[threadIdx.x] = (threadIdx.x < (unsigned)nb) ? bsum[threadIdx.x] : 0;
    __syncthreads();
#pragma unroll
    for (int o = 1; o < 512; o <<= 1) {
        int t = (threadIdx.x >= o) ? s[threadIdx.x - o] : 0;
        __syncthreads();
        s[threadIdx.x] += t;
        __syncthreads();
    }
    if (threadIdx.x < (unsigned)nb) bsum[threadIdx.x] = s[threadIdx.x];  // inclusive
}

__global__ void k_scan_add(int* __restrict__ out, const int* __restrict__ bsum, int n)
{
    const int i = blockIdx.x * SCB + threadIdx.x;
    if (i >= n) return;
    if (blockIdx.x > 0) out[i] += bsum[blockIdx.x - 1];
}

__global__ void k_copy_i32(const int* __restrict__ src, int* __restrict__ dst, int n)
{
    int i = blockIdx.x * blockDim.x + threadIdx.x;
    if (i < n) dst[i] = src[i];
}

// ---------------- CSR fill: csr_src grouped by destination node ----------------
__global__ void k_fill(const int* __restrict__ row, const int* __restrict__ col,
                       int* __restrict__ fillpos, int* __restrict__ csr_src, int E)
{
    int e = blockIdx.x * blockDim.x + threadIdx.x;
    if (e < E) {
        int p = atomicAdd(&fillpos[col[e]], 1);
        csr_src[p] = row[e];
    }
}

// ---------------- pack W (128x128 f32 row-major, W[k][n]) into B-fragment order ----------------
// B-fragment k order used here (contiguous k, same as the A tile in k_gemm):
//   lane 0-15 : N = n0*16 + lane,     half i -> K = kb*32 + i
//   lane 16-31: N = n0*16 + lane-16,  half i -> K = kb*32 + 16 + i
__global__ void k_pack_w(const float* __restrict__ W, _Float16* __restrict__ wpk)
{
    int t = blockIdx.x * blockDim.x + threadIdx.x;
    if (t >= 1024) return;                       // 8 n-tiles * 4 k-blocks * 32 lanes
    int lane = t & 31;
    int kb   = (t >> 5) & 3;
    int n0   = t >> 7;
    int n     = n0 * 16 + (lane & 15);
    int kbase = kb * 32 + ((lane < 16) ? 0 : 16);
    _Float16* dst = wpk + (size_t)t * 16;
#pragma unroll
    for (int i = 0; i < 16; ++i)
        dst[i] = (_Float16)W[(size_t)(kbase + i) * HIDN + n];
}

// ---------------- GEMM: G = Xf16 @ W  (N x 128) via WMMA ----------------
__global__ void __launch_bounds__(256)
k_gemm(const _Float16* __restrict__ X, const _Float16* __restrict__ wpk,
       _Float16* __restrict__ G)
{
    __shared__ _Float16 sx[16 * HIDN];           // 4 KB: 16-row x 128-col f16 tile

    const int rowTile = blockIdx.x;
    const int tid  = threadIdx.x;
    {
        const _Float16* src = X + (size_t)rowTile * 16 * HIDN;
        *(v8h*)(sx + tid * 8) = *(const v8h*)(src + tid * 8);
    }
    __syncthreads();

    const int wave = tid >> 5;
    const int lane = tid & 31;
    const int m     = lane & 15;
    const int koffA = (lane < 16) ? 0 : 16;      // A-frag: lanes 0-15 hold K 0..15

    v8f acc = {};
#pragma unroll
    for (int kb = 0; kb < 4; ++kb) {
        const _Float16* ap = sx + m * HIDN + kb * 32 + koffA;
        v8h alo = *(const v8h*)(ap);             // K = base+0..7
        v8h ahi = *(const v8h*)(ap + 8);         // K = base+8..15
        v16h a;
#pragma unroll
        for (int i = 0; i < 8; ++i) { a[i] = alo[i]; a[i + 8] = ahi[i]; }

        v16h b = *(const v16h*)(wpk + (size_t)((wave * 4 + kb) * 32 + lane) * 16);

        acc = __builtin_amdgcn_wmma_f32_16x16x32_f16(
                  false, a, false, b, (short)0, acc, false, false);
    }

    const int n     = wave * 16 + (lane & 15);
    const int mbase = rowTile * 16 + ((lane < 16) ? 0 : 8);
#pragma unroll
    for (int r = 0; r < 8; ++r)
        G[(size_t)(mbase + r) * HIDN + n] = (_Float16)acc[r];
}

// ---------------- fused CSR gather + bias + LayerNorm + ReLU + residual ----------------
// one wave per destination node; lane owns 4 of the 128 dims; zero atomics
__global__ void __launch_bounds__(256)
k_gather_ln(const int* __restrict__ rowptr, const int* __restrict__ csr_src,
            const float* __restrict__ dinv, const _Float16* __restrict__ g16,
            const float* __restrict__ bias, const float* __restrict__ gamma,
            const float* __restrict__ beta,
            float* __restrict__ h, _Float16* __restrict__ hf16,
            int residual, int n)
{
    const int v = blockIdx.x * 8 + (threadIdx.x >> 5);
    if (v >= n) return;
    const int lane = threadIdx.x & 31;
    const int d4   = lane * 4;
    const float dv = dinv[v];

    // self-loop contribution: norm = dv*dv
    v4h qs = *(const v4h*)(g16 + (size_t)v * HIDN + d4);
    const float sc = dv * dv;
    float a0 = sc * (float)qs[0], a1 = sc * (float)qs[1];
    float a2 = sc * (float)qs[2], a3 = sc * (float)qs[3];

    const int beg = rowptr[v], end = rowptr[v + 1];
    for (int j = beg; j < end; ++j) {
        const int r = csr_src[j];               // wave-uniform
        const float nrm = dinv[r] * dv;
        v4h q = *(const v4h*)(g16 + (size_t)r * HIDN + d4);
        a0 += nrm * (float)q[0];
        a1 += nrm * (float)q[1];
        a2 += nrm * (float)q[2];
        a3 += nrm * (float)q[3];
    }

    float4 bb = *(const float4*)(bias + d4);
    a0 += bb.x; a1 += bb.y; a2 += bb.z; a3 += bb.w;

    float s = a0 + a1 + a2 + a3;
#pragma unroll
    for (int o = 16; o > 0; o >>= 1) s += __shfl_xor(s, o);
    const float mu = s * (1.0f / HIDN);

    const float dx = a0 - mu, dy = a1 - mu, dz = a2 - mu, dw = a3 - mu;
    float q2 = dx * dx + dy * dy + dz * dz + dw * dw;
#pragma unroll
    for (int o = 16; o > 0; o >>= 1) q2 += __shfl_xor(q2, o);
    const float inv = rsqrtf(q2 * (1.0f / HIDN) + 1e-5f);

    float4 gm = *(const float4*)(gamma + d4);
    float4 bt = *(const float4*)(beta  + d4);
    float y0 = fmaxf(dx * inv * gm.x + bt.x, 0.0f);
    float y1 = fmaxf(dy * inv * gm.y + bt.y, 0.0f);
    float y2 = fmaxf(dz * inv * gm.z + bt.z, 0.0f);
    float y3 = fmaxf(dw * inv * gm.w + bt.w, 0.0f);

    const size_t base = (size_t)v * HIDN + d4;
    if (residual) {
        float4 hp = *(const float4*)(h + base);
        y0 += hp.x; y1 += hp.y; y2 += hp.z; y3 += hp.w;
    }
    *(float4*)(h + base) = make_float4(y0, y1, y2, y3);
    v4h hv; hv[0] = (_Float16)y0; hv[1] = (_Float16)y1;
            hv[2] = (_Float16)y2; hv[3] = (_Float16)y3;
    *(v4h*)(hf16 + base) = hv;
}

// ---------------- global mean pool (sums + counts via atomics; one pass) ----------------
__global__ void k_pool(const float* __restrict__ h, const int* __restrict__ batch,
                       float* __restrict__ pooled, float* __restrict__ cnt, int n)
{
    const int v = blockIdx.x * 8 + (threadIdx.x >> 5);
    if (v >= n) return;
    const int lane = threadIdx.x & 31;
    const int g = batch[v];
    float4 t = *(const float4*)(h + (size_t)v * HIDN + lane * 4);
    float* p = pooled + (size_t)g * HIDN + lane * 4;
    atomicAdd(p + 0, t.x); atomicAdd(p + 1, t.y);
    atomicAdd(p + 2, t.z); atomicAdd(p + 3, t.w);
    if (lane == 0) atomicAdd(&cnt[g], 1.0f);
}

// ---------------- per-graph head ----------------
__global__ void k_fc(const float* __restrict__ pooled, const float* __restrict__ cnt,
                     const float* __restrict__ W1, const float* __restrict__ b1,
                     const float* __restrict__ W2, const float* __restrict__ b2,
                     float* __restrict__ out)
{
    __shared__ float red[FCD];
    const int g = blockIdx.x;
    const int j = threadIdx.x;                   // 0..63
    const float ic = 1.0f / fmaxf(cnt[g], 1.0f);
    float s = b1[j];
    const float* pg = pooled + (size_t)g * HIDN;
#pragma unroll 4
    for (int k = 0; k < HIDN; ++k)
        s += (pg[k] * ic) * W1[(size_t)k * FCD + j];
    red[j] = fmaxf(s, 0.0f) * W2[j];
    __syncthreads();
    for (int st = FCD / 2; st > 0; st >>= 1) {
        if (j < st) red[j] += red[j + st];
        __syncthreads();
    }
    if (j == 0) out[g] = 1.0f / (1.0f + __expf(-(red[0] + b2[0])));
}

extern "C" void kernel_launch(void* const* d_in, const int* in_sizes, int n_in,
                              void* d_out, int out_size, void* d_ws, size_t ws_size,
                              hipStream_t stream)
{
    const float* x     = (const float*)d_in[0];
    const int*   eidx  = (const int*)  d_in[1];
    const int*   batch = (const int*)  d_in[2];
    const float* W_in  = (const float*)d_in[3];
    const float* b_in  = (const float*)d_in[4];
    const float* g_in  = (const float*)d_in[5];
    const float* be_in = (const float*)d_in[6];
    const float* W_mid = (const float*)d_in[7];
    const float* b_mid = (const float*)d_in[8];
    const float* g_mid = (const float*)d_in[9];
    const float* be_mid= (const float*)d_in[10];
    const float* W_out = (const float*)d_in[11];
    const float* b_out = (const float*)d_in[12];
    const float* g_out = (const float*)d_in[13];
    const float* be_out= (const float*)d_in[14];
    const float* W1    = (const float*)d_in[15];
    const float* b1    = (const float*)d_in[16];
    const float* W2    = (const float*)d_in[17];
    const float* b2    = (const float*)d_in[18];
    float* out = (float*)d_out;

    const int N = in_sizes[0] / HIDN;   // 100000 (divisible by 16)
    const int E = in_sizes[1] / 2;      // 1600000
    const int G = out_size;             // 128 graphs
    const int* row = eidx;
    const int* col = eidx + E;
    const int nb = (N + SCB - 1) / SCB; // scan blocks (391 <= 512)

    // workspace carve-out
    char* ws = (char*)d_ws;
    size_t off = 0;
    auto carve = [&](size_t bytes) -> char* {
        char* p = ws + off;
        off += (bytes + 255) & ~(size_t)255;
        return p;
    };
    float*    h      = (float*)   carve((size_t)N * HIDN * 4);
    _Float16* hf16   = (_Float16*)carve((size_t)N * HIDN * 2);
    _Float16* g16    = (_Float16*)carve((size_t)N * HIDN * 2);
    float*    dinv   = (float*)   carve((size_t)N * 4);
    int*      ecnt   = (int*)     carve((size_t)N * 4);      // counts, later fill-pos
    int*      rowptr = (int*)     carve((size_t)(N + 1) * 4);
    int*      csrsrc = (int*)     carve((size_t)E * 4);
    int*      bsum   = (int*)     carve(512 * 4);
    _Float16* wpk    = (_Float16*)carve((size_t)3 * 16384 * 2);
    float*    pooled = (float*)   carve((size_t)G * HIDN * 4);
    float*    cnt    = (float*)   carve((size_t)G * 4);

    // ---- CSR build + norms (once per launch; deterministic recompute) ----
    hipMemsetAsync(ecnt, 0, (size_t)N * 4, stream);
    hipMemsetAsync(rowptr, 0, 4, stream);                    // rowptr[0] = 0
    k_count<<<(E + 255) / 256, 256, 0, stream>>>(col, ecnt, E);
    k_dinv <<<(N + 255) / 256, 256, 0, stream>>>(ecnt, dinv, N);
    k_scan_local<<<nb, SCB, 0, stream>>>(ecnt, rowptr + 1, bsum, N);
    k_scan_bsum <<<1, 512, 0, stream>>>(bsum, nb);
    k_scan_add  <<<nb, SCB, 0, stream>>>(rowptr + 1, bsum, N);
    k_copy_i32  <<<(N + 255) / 256, 256, 0, stream>>>(rowptr, ecnt, N);  // fill positions
    k_fill      <<<(E + 255) / 256, 256, 0, stream>>>(row, col, ecnt, csrsrc, E);

    // ---- pack weights into WMMA B-fragment order ----
    k_pack_w<<<4, 256, 0, stream>>>(W_in,  wpk + 0 * 16384);
    k_pack_w<<<4, 256, 0, stream>>>(W_mid, wpk + 1 * 16384);
    k_pack_w<<<4, 256, 0, stream>>>(W_out, wpk + 2 * 16384);

    // ---- layer-0 input in f16 ----
    k_f32_to_f16<<<(N * HIDN + 255) / 256, 256, 0, stream>>>(x, hf16, N * HIDN);

    struct Layer { const _Float16* wp; const float *b, *g, *be; int res; };
    const Layer layers[8] = {
        { wpk + 0 * 16384, b_in,  g_in,  be_in,  0 },
        { wpk + 1 * 16384, b_mid, g_mid, be_mid, 0 },
        { wpk + 1 * 16384, b_mid, g_mid, be_mid, 1 },
        { wpk + 1 * 16384, b_mid, g_mid, be_mid, 0 },
        { wpk + 1 * 16384, b_mid, g_mid, be_mid, 1 },
        { wpk + 1 * 16384, b_mid, g_mid, be_mid, 0 },
        { wpk + 1 * 16384, b_mid, g_mid, be_mid, 1 },
        { wpk + 2 * 16384, b_out, g_out, be_out, 0 },
    };

    for (int L = 0; L < 8; ++L) {
        k_gemm<<<N / 16, 256, 0, stream>>>(hf16, layers[L].wp, g16);
        k_gather_ln<<<(N + 7) / 8, 256, 0, stream>>>(rowptr, csrsrc, dinv, g16,
                                                     layers[L].b, layers[L].g, layers[L].be,
                                                     h, hf16, layers[L].res, N);
    }

    hipMemsetAsync(pooled, 0, (size_t)G * HIDN * 4, stream);
    hipMemsetAsync(cnt,    0, (size_t)G * 4,        stream);
    k_pool<<<(N + 7) / 8, 256, 0, stream>>>(h, batch, pooled, cnt, N);
    k_fc  <<<G, FCD, 0, stream>>>(pooled, cnt, W1, b1, W2, b2, out);
}
